// Encoder_21328807592584
// MI455X (gfx1250) — compile-verified, matrix operand order checked by reading, not run on hardware
//
#include <hip/hip_runtime.h>
#include <hip/hip_bf16.h>

// ---------------------------------------------------------------------------
// CDNA5 (gfx1250) wave32 WMMA implementation of the graph-VAE encoder.
//  - conv stack as chained im2col GEMMs (stride-2 2x2 conv = pure permutation,
//    so each layer stores directly into the next layer's im2col layout)
//  - fragments loaded straight from global memory as 2x global_load_b128 per
//    operand (no LDS staging): A row-major MxK, B stored transposed (N x K)
//  - v_wmma_f32_16x16x32_f16 with f32 accumulation everywhere
// ---------------------------------------------------------------------------

typedef __attribute__((ext_vector_type(16))) _Float16 v16h;
typedef __attribute__((ext_vector_type(8)))  _Float16 v8h;
typedef __attribute__((ext_vector_type(8)))  float    v8f;

#define NIMG 400      // BS*G = 16*25
#define FEAT 512
#define ZD   128

// ---------------------------------------------------------------------------
// Load one 16x32 f16 fragment whose 16 rows start at `base` with row stride
// `ldk` halfs. Per CDNA5 ISA: lane L -> row r = L%16, halves
// [kh, kh+8) and [16+kh, 16+kh+8) with kh = (L/16)*8. Each chunk is a
// contiguous 16-byte global_load_b128. Works for A (rows = m) and for
// B^T (rows = n) identically.
// ---------------------------------------------------------------------------
__device__ __forceinline__ v16h frag_rows(const _Float16* __restrict__ base, int ldk) {
    const int lane = threadIdx.x & 31;
    const int r  = lane & 15;
    const int kh = (lane >> 4) << 3;
    const _Float16* p = base + (size_t)r * ldk + kh;
    const v8h lo = *(const v8h*)(p);
    const v8h hi = *(const v8h*)(p + 16);
    v16h f;
#pragma unroll
    for (int q = 0; q < 8; ++q) { f[q] = lo[q]; f[8 + q] = hi[q]; }
    return f;
}

// ---------------------------------------------------------------------------
// Layer 0: 1->8 channels, 2x2/stride2 on (400,256,256) f32. Bandwidth-bound
// (K=4); VALU kernel that writes its output directly as the im2col matrix
// A1 (400*64*64 rows, K=32) for layer 1.
// ---------------------------------------------------------------------------
__global__ void conv_layer0(const float* __restrict__ X,
                            const float* __restrict__ W0,   // (8,1,2,2) = (8,4)
                            const float* __restrict__ B0,
                            _Float16* __restrict__ A1) {
    int p = blockIdx.x * blockDim.x + threadIdx.x;      // 400*128*128 pixels
    if (p >= NIMG * 128 * 128) return;
    const int ni  = p >> 14;
    const int rem = p & 16383;
    const int oy  = rem >> 7;
    const int ox  = rem & 127;
    const float* q = X + ((size_t)ni * 256 + oy * 2) * 256 + ox * 2;
    const float x00 = q[0], x01 = q[1], x10 = q[256], x11 = q[257];
    const int row = (ni << 12) + ((oy >> 1) << 6) + (ox >> 1);   // next im2col row
    const int pos = ((oy & 1) << 1) + (ox & 1);
    _Float16* dst = A1 + (size_t)row * 32 + pos;
#pragma unroll
    for (int c = 0; c < 8; ++c) {
        float v = B0[c] + W0[c * 4 + 0] * x00 + W0[c * 4 + 1] * x01
                        + W0[c * 4 + 2] * x10 + W0[c * 4 + 3] * x11;
        v = fmaxf(v, 0.f);
        dst[c * 4] = (_Float16)v;
    }
}

// ---------------------------------------------------------------------------
// Conv layers 1..6 as NT GEMM: A (M x K im2col, row-major f16) x Wt (N x K,
// native conv_w layout!) -> store into NEXT layer's im2col matrix with fused
// bias(+ReLU). One wave per block; MT=4 m-tiles share each B fragment.
// M = 400*OH*OH (multiple of 64), N = Cout (mult of 16), K = 4*Cin (mult 32).
// ---------------------------------------------------------------------------
template <int MT>
__global__ void convgemm_wmma(const _Float16* __restrict__ A,
                              const _Float16* __restrict__ Wt,
                              const float* __restrict__ bias,
                              _Float16* __restrict__ Anext,
                              int K, int N, int OH, int doRelu) {
    const int lane = threadIdx.x;
    const int m0 = blockIdx.x * (16 * MT);
    const int n0 = blockIdx.y * 16;
    v8f acc[MT] = {};

    for (int k0 = 0; k0 < K; k0 += 32) {
        const v16h bf = frag_rows(Wt + (size_t)n0 * K + k0, K);
        if (k0 + 32 < K)
            __builtin_prefetch(Wt + (size_t)n0 * K + k0 + 32, 0, 1);  // global_prefetch_b8
#pragma unroll
        for (int t = 0; t < MT; ++t) {
            const v16h af = frag_rows(A + (size_t)(m0 + t * 16) * K + k0, K);
            acc[t] = __builtin_amdgcn_wmma_f32_16x16x32_f16(false, af, false, bf,
                                                            (short)0, acc[t], false, false);
        }
    }

#if __has_builtin(__builtin_amdgcn_s_wait_tensorcnt)
    __builtin_amdgcn_s_wait_tensorcnt(0);   // no-op; TDM staging hook
#endif

    const int half = lane >> 4;
    const int n = n0 + (lane & 15);
    const float bv = bias[n];
    const int OHW = OH * OH;
    const int OH2 = OH >> 1;
    const int ldn = N * 4;
#pragma unroll
    for (int t = 0; t < MT; ++t) {
#pragma unroll
        for (int r = 0; r < 8; ++r) {
            const int m = m0 + t * 16 + r + half * 8;
            float v = acc[t][r] + bv;
            if (doRelu) v = fmaxf(v, 0.f);
            const int ni  = m / OHW;
            const int rem = m - ni * OHW;
            const int oy  = rem / OH;
            const int ox  = rem - oy * OH;
            const int row = (ni * OH2 + (oy >> 1)) * OH2 + (ox >> 1);
            const int col = n * 4 + ((oy & 1) << 1) + (ox & 1);
            Anext[(size_t)row * ldn + col] = (_Float16)v;
        }
    }
}

// ---------------------------------------------------------------------------
// Generic NT GEMM: C = act(A @ B^T + bias). A: MxK f16; Bt: NxK f16.
// Used for aggregator (MT=5, M=400) and mu/logvar heads (MT=1, M=16).
// ---------------------------------------------------------------------------
template <int MT>
__global__ void gemm_nt_wmma(const _Float16* __restrict__ A,
                             const _Float16* __restrict__ Bt,
                             const float* __restrict__ bias,
                             float* __restrict__ Cf, _Float16* __restrict__ Ch,
                             int N, int K, int doRelu) {
    const int lane = threadIdx.x;
    const int m0 = blockIdx.x * (16 * MT);
    const int n0 = blockIdx.y * 16;
    v8f acc[MT] = {};

    for (int k0 = 0; k0 < K; k0 += 32) {
        const v16h bf = frag_rows(Bt + (size_t)n0 * K + k0, K);
        if (k0 + 32 < K)
            __builtin_prefetch(Bt + (size_t)n0 * K + k0 + 32, 0, 1);
#pragma unroll
        for (int t = 0; t < MT; ++t) {
            const v16h af = frag_rows(A + (size_t)(m0 + t * 16) * K + k0, K);
            acc[t] = __builtin_amdgcn_wmma_f32_16x16x32_f16(false, af, false, bf,
                                                            (short)0, acc[t], false, false);
        }
    }

    const int half = lane >> 4;
    const int n = n0 + (lane & 15);
    const float bv = bias ? bias[n] : 0.f;
#pragma unroll
    for (int t = 0; t < MT; ++t) {
#pragma unroll
        for (int r = 0; r < 8; ++r) {
            const int m = m0 + t * 16 + r + half * 8;
            float v = acc[t][r] + bv;
            if (doRelu) v = fmaxf(v, 0.f);
            if (Cf) Cf[(size_t)m * N + n] = v;
            if (Ch) Ch[(size_t)m * N + n] = (_Float16)v;
        }
    }
}

// ---------------------------------------------------------------------------
// Weight conversion helpers
// ---------------------------------------------------------------------------
__global__ void f32_to_f16_kernel(const float* __restrict__ src,
                                  _Float16* __restrict__ dst, int n) {
    int i = blockIdx.x * blockDim.x + threadIdx.x;
    if (i < n) dst[i] = (_Float16)src[i];
}

// (K,N) f32 -> (N,K) f16   (so the GEMM sees B transposed, K-contiguous)
__global__ void f32_to_f16_transpose(const float* __restrict__ src,
                                     _Float16* __restrict__ dst, int K, int N) {
    int idx = blockIdx.x * blockDim.x + threadIdx.x;
    if (idx >= K * N) return;
    const int k = idx / N, n = idx - k * N;
    dst[(size_t)n * K + k] = (_Float16)src[idx];
}

// ---------------------------------------------------------------------------
// cn1 (400,512) = 2x2 max-pool of layer-6 output stored as (400, 512*4)
// ---------------------------------------------------------------------------
__global__ void maxpool_final(const _Float16* __restrict__ x, float* __restrict__ cn1) {
    int idx = blockIdx.x * blockDim.x + threadIdx.x;
    if (idx >= NIMG * FEAT) return;
    const _Float16* q = x + (size_t)idx * 4;   // (ni*512 + c)*4 + j
    float v = fmaxf(fmaxf((float)q[0], (float)q[1]), fmaxf((float)q[2], (float)q[3]));
    cn1[idx] = v;
}

// ---------------------------------------------------------------------------
// Per-batch graph coefficients (one 256-thread block per batch, LDS-staged)
// ---------------------------------------------------------------------------
__global__ void graph_coeff_kernel(const float* __restrict__ cn1,
                                   const float* __restrict__ masks,   // (16,24,24)
                                   const int* __restrict__ rel_dist,  // (16,24,24)
                                   const float* __restrict__ w_rel,   // (20,512)
                                   float* __restrict__ coff_adjs,     // (16,25,25)
                                   float* __restrict__ term2) {       // (16,25)
    const int b = blockIdx.x;
    const int t = threadIdx.x;
    __shared__ float part[24][FEAT];
    __shared__ float coff[24][24];
    __shared__ float rp[24][24];
    __shared__ float adjs[25][25];
    __shared__ float nrm[24];

    for (int idx = t; idx < 24 * FEAT; idx += 256) {
        const int i = idx >> 9, d = idx & 511;
        part[i][d] = cn1[((size_t)b * 25 + 1 + i) * FEAT + d];
    }
    __syncthreads();

    if (t < 24) {
        float s = 0.f;
        for (int d = 0; d < FEAT; ++d) { float v = part[t][d]; s += v * v; }
        nrm[t] = sqrtf(s);
    }
    __syncthreads();

    for (int idx = t; idx < 576; idx += 256) {
        const int i = idx / 24, j = idx % 24;
        float s = 0.f;
        for (int d = 0; d < FEAT; ++d) s += part[i][d] * part[j][d];
        coff[i][j] = s / (nrm[i] * nrm[j] + 1e-10f);
    }
    for (int idx = t; idx < 576; idx += 256) {
        const int i = idx / 24, j = idx % 24;
        const float* e = w_rel + (size_t)rel_dist[((size_t)b * 24 + i) * 24 + j] * FEAT;
        float s = 0.f;
        for (int d = 0; d < FEAT; ++d) s += part[i][d] * e[d];
        rp[i][j] = tanhf(s);
    }
    __syncthreads();

    if (t < 24) {
        const int i = t;
        int i1 = 0; float m1 = -1e30f;
        for (int j = 0; j < 24; ++j) { float v = coff[i][j]; if (v > m1) { m1 = v; i1 = j; } }
        int i2 = 0; float m2 = -1e30f;
        for (int j = 0; j < 24; ++j) { float v = (j == i1) ? 0.f : coff[i][j]; if (v > m2) { m2 = v; i2 = j; } }
        int i3 = 0; float m3 = -1e30f;
        for (int j = 0; j < 24; ++j) { float v = (j == i1 || j == i2) ? 0.f : coff[i][j]; if (v > m3) { m3 = v; i3 = j; } }
        float row[25];
        row[0] = 0.5f * masks[((size_t)b * 24 + i) * 24 + 0];
        float sum = row[0];
        for (int j = 0; j < 24; ++j) {
            const float w = (j == i1) ? 1.f : (j == i2) ? 0.5f : (j == i3) ? 0.2f : 0.f;
            const float v = coff[i][j] * w * masks[((size_t)b * 24 + i) * 24 + j];
            row[j + 1] = v; sum += v;
        }
        const float inv = 1.f / (sum + 1e-10f);
        for (int j = 0; j < 25; ++j) adjs[i + 1][j] = row[j] * inv;
    } else if (t == 24) {
        adjs[0][0] = 1.f / (1.f + 1e-10f);
        for (int j = 1; j < 25; ++j) adjs[0][j] = 0.f;
    }
    __syncthreads();

    if (t < 24) {
        const int i = t;
        float s = 0.f;
        for (int j = 0; j < 24; ++j) s += adjs[i + 1][j + 1] * rp[i][j];
        term2[b * 25 + i + 1] = s;
    } else if (t == 24) {
        term2[b * 25 + 0] = 0.f;
    }
    for (int idx = t; idx < 625; idx += 256)
        coff_adjs[(size_t)b * 625 + idx] = adjs[idx / 25][idx % 25];
}

// ---------------------------------------------------------------------------
// BN1 stats / apply
// ---------------------------------------------------------------------------
__global__ void bn_stats_400(const float* __restrict__ cn1,
                             float* __restrict__ mean, float* __restrict__ invstd) {
    int c = blockIdx.x * blockDim.x + threadIdx.x;
    if (c >= FEAT) return;
    float s = 0.f, s2 = 0.f;
    for (int r = 0; r < NIMG; ++r) {
        float v = fmaxf(cn1[(size_t)r * FEAT + c], 0.f);
        s += v; s2 += v * v;
    }
    const float m = s / (float)NIMG;
    mean[c] = m;
    invstd[c] = rsqrtf(s2 / (float)NIMG - m * m + 1e-5f);
}

__global__ void bn1_apply(const float* __restrict__ cn1,
                          const float* __restrict__ mean, const float* __restrict__ invstd,
                          const float* __restrict__ g, const float* __restrict__ b,
                          float* __restrict__ bn1) {
    int idx = blockIdx.x * blockDim.x + threadIdx.x;
    if (idx >= NIMG * FEAT) return;
    const int c = idx & 511;
    const float v = fmaxf(cn1[idx], 0.f);
    bn1[idx] = (v - mean[c]) * invstd[c] * g[c] + b[c];
}

// ---------------------------------------------------------------------------
// cn2 = coff_adjs @ bn1 + term2 + term3
// ---------------------------------------------------------------------------
__global__ void cn2_kernel(const float* __restrict__ adjs, const float* __restrict__ bn1,
                           const float* __restrict__ term2, const float* __restrict__ abs_pe,
                           const float* __restrict__ masks,
                           float* __restrict__ cn2f, _Float16* __restrict__ cn2h) {
    const int bi = blockIdx.x;             // 0..399
    const int b = bi / 25, i = bi % 25;
    const int d = threadIdx.x;
    float s = 0.f;
    for (int j = 0; j < 25; ++j)
        s += adjs[((size_t)b * 25 + i) * 25 + j] * bn1[((size_t)b * 25 + j) * FEAT + d];
    const float t3 = (i == 0)
        ? -1.1f
        : abs_pe[((size_t)b * 24 + (i - 1)) * FEAT + d] * masks[((size_t)b * 24 + (i - 1)) * 24 + 0];
    const float v = s + term2[b * 25 + i] + t3;
    cn2f[(size_t)bi * FEAT + d] = v;
    cn2h[(size_t)bi * FEAT + d] = (_Float16)v;
}

__global__ void gcn_kernel(const float* __restrict__ a2, const float* __restrict__ cn2f,
                           float* __restrict__ gcn) {
    const int b = blockIdx.x, d = threadIdx.x;
    float s = 0.f;
    for (int i = 0; i < 25; ++i) {
        const size_t idx = ((size_t)b * 25 + i) * FEAT + d;
        s += a2[idx] + cn2f[idx];
    }
    gcn[(size_t)b * FEAT + d] = s;
}

__global__ void bn2_kernel(const float* __restrict__ gcn,
                           const float* __restrict__ g, const float* __restrict__ b,
                           _Float16* __restrict__ bn2h) {
    const int d = threadIdx.x;
    float s = 0.f, s2 = 0.f;
    for (int i = 0; i < 16; ++i) { float v = gcn[(size_t)i * FEAT + d]; s += v; s2 += v * v; }
    const float m = s / 16.f;
    const float inv = rsqrtf(s2 / 16.f - m * m + 1e-5f);
    for (int i = 0; i < 16; ++i) {
        const float t = tanhf((gcn[(size_t)i * FEAT + d] - m) * inv * g[d] + b[d]);
        bn2h[(size_t)i * FEAT + d] = (_Float16)t;
    }
}

__global__ void z_kernel(const float* __restrict__ mu, const float* __restrict__ lv,
                         const float* __restrict__ eps, float* __restrict__ z) {
    int idx = blockIdx.x * blockDim.x + threadIdx.x;
    if (idx < 16 * ZD) z[idx] = mu[idx] + expf(0.5f * lv[idx]) * eps[idx];
}

// ---------------------------------------------------------------------------
// Host pipeline
// ---------------------------------------------------------------------------
extern "C" void kernel_launch(void* const* d_in, const int* in_sizes, int n_in,
                              void* d_out, int out_size, void* d_ws, size_t ws_size,
                              hipStream_t stream) {
    (void)in_sizes; (void)n_in; (void)out_size; (void)ws_size;
    const float* in_graphs = (const float*)d_in[0];
    const float* in_masks  = (const float*)d_in[1];
    const int*   rel_dist  = (const int*)  d_in[2];
    const float* abs_pe    = (const float*)d_in[3];
    const float* eps       = (const float*)d_in[4];
    const float* conv_w[7]; const float* conv_b[7];
    for (int i = 0; i < 7; ++i) { conv_w[i] = (const float*)d_in[5 + i]; conv_b[i] = (const float*)d_in[12 + i]; }
    const float* bn1_g   = (const float*)d_in[19];
    const float* bn1_b   = (const float*)d_in[20];
    const float* aggr_w1 = (const float*)d_in[21];
    const float* aggr_b1 = (const float*)d_in[22];
    const float* aggr_w2 = (const float*)d_in[23];
    const float* aggr_b2 = (const float*)d_in[24];
    const float* bn2_g   = (const float*)d_in[25];
    const float* bn2_b   = (const float*)d_in[26];
    const float* w_rel   = (const float*)d_in[27];
    const float* mu_w    = (const float*)d_in[28];
    const float* mu_b    = (const float*)d_in[29];
    const float* lv_w    = (const float*)d_in[30];
    const float* lv_b    = (const float*)d_in[31];
    float* out = (float*)d_out;

    // ---- workspace carve-out ----
    char* p = (char*)d_ws;
    auto alloc = [&](size_t bytes) -> char* {
        char* r = p; p += (bytes + 255) & ~(size_t)255; return r;
    };
    static const int CHc[8] = {1, 8, 32, 64, 128, 256, 512, 512};
    const size_t ACT = 52428800ULL;                       // max im2col elems (f16)
    _Float16* actA = (_Float16*)alloc(ACT * 2);
    _Float16* actB = (_Float16*)alloc(ACT * 2);
    _Float16* wk[7];
    for (int i = 1; i < 7; ++i) wk[i] = (_Float16*)alloc((size_t)CHc[i + 1] * CHc[i] * 4 * 2);
    _Float16* w1t  = (_Float16*)alloc(512 * 512 * 2);     // aggr_w1^T (N,K)
    _Float16* w2t  = (_Float16*)alloc(512 * 512 * 2);
    _Float16* muwt = (_Float16*)alloc(128 * 512 * 2);     // mu_w^T
    _Float16* lvwt = (_Float16*)alloc(128 * 512 * 2);
    float* cn1    = (float*)alloc((size_t)NIMG * FEAT * 4);
    float* adjs   = (float*)alloc(16 * 625 * 4);
    float* term2  = (float*)alloc(16 * 25 * 4);
    float* mean1  = (float*)alloc(FEAT * 4);
    float* inv1   = (float*)alloc(FEAT * 4);
    float* bn1    = (float*)alloc((size_t)NIMG * FEAT * 4);
    float* cn2f   = (float*)alloc((size_t)NIMG * FEAT * 4);
    _Float16* cn2h = (_Float16*)alloc((size_t)NIMG * FEAT * 2);
    _Float16* a1h  = (_Float16*)alloc((size_t)NIMG * FEAT * 2);
    float* a2f    = (float*)alloc((size_t)NIMG * FEAT * 4);
    float* gcn    = (float*)alloc(16 * FEAT * 4);
    _Float16* bn2h = (_Float16*)alloc(16 * FEAT * 2);

    // ---- weight conversions ----
    for (int i = 1; i < 7; ++i) {
        const int n = CHc[i + 1] * CHc[i] * 4;            // native (Cout,K) layout
        f32_to_f16_kernel<<<(n + 255) / 256, 256, 0, stream>>>(conv_w[i], wk[i], n);
    }
    f32_to_f16_transpose<<<(512 * 512 + 255) / 256, 256, 0, stream>>>(aggr_w1, w1t, 512, 512);
    f32_to_f16_transpose<<<(512 * 512 + 255) / 256, 256, 0, stream>>>(aggr_w2, w2t, 512, 512);
    f32_to_f16_transpose<<<(512 * 128 + 255) / 256, 256, 0, stream>>>(mu_w, muwt, 512, 128);
    f32_to_f16_transpose<<<(512 * 128 + 255) / 256, 256, 0, stream>>>(lv_w, lvwt, 512, 128);

    // ---- conv stack ----
    // Layer 0 (f32 in -> im2col A1 f16)
    conv_layer0<<<(NIMG * 128 * 128 + 255) / 256, 256, 0, stream>>>(in_graphs, conv_w[0],
                                                                    conv_b[0], actA);
    // Layers 1..6: NT WMMA GEMM, output written as next layer's im2col matrix
    _Float16* cur = actA; _Float16* nxt = actB;
    for (int i = 1; i < 7; ++i) {
        const int OH = 128 >> i;                          // this layer's output spatial
        const int M  = NIMG * OH * OH * 4;                // rows of A_i = 400*(2*OH)^2/4... 
        // rows of A_i: one per output pixel of this layer = NIMG*OH*OH
        const int Mi = NIMG * OH * OH;
        const int K  = CHc[i] * 4;
        const int N  = CHc[i + 1];
        (void)M;
        dim3 grid(Mi / 64, N / 16);                       // MT=4 -> 64 rows/block
        convgemm_wmma<4><<<grid, 32, 0, stream>>>(cur, wk[i], conv_b[i], nxt,
                                                  K, N, OH, (i < 6) ? 1 : 0);
        _Float16* t = cur; cur = nxt; nxt = t;
    }
    // cur = A7 = (400, 2048): row ni, col c*4 + (2x2 patch pos)

    maxpool_final<<<(NIMG * FEAT + 255) / 256, 256, 0, stream>>>(cur, cn1);
    graph_coeff_kernel<<<16, 256, 0, stream>>>(cn1, in_masks, rel_dist, w_rel, adjs, term2);
    bn_stats_400<<<2, 256, 0, stream>>>(cn1, mean1, inv1);
    bn1_apply<<<(NIMG * FEAT + 255) / 256, 256, 0, stream>>>(cn1, mean1, inv1, bn1_g, bn1_b, bn1);
    cn2_kernel<<<NIMG, FEAT, 0, stream>>>(adjs, bn1, term2, abs_pe, in_masks, cn2f, cn2h);

    // ---- aggregator GEMMs: (400x512)@(512x512) twice, MT=5 (400 = 5*5*16) ----
    gemm_nt_wmma<5><<<dim3(5, 32), 32, 0, stream>>>(cn2h, w1t, aggr_b1,
                                                    nullptr, a1h, FEAT, FEAT, 1);
    gemm_nt_wmma<5><<<dim3(5, 32), 32, 0, stream>>>(a1h, w2t, aggr_b2,
                                                    a2f, nullptr, FEAT, FEAT, 1);

    gcn_kernel<<<16, FEAT, 0, stream>>>(a2f, cn2f, gcn);
    bn2_kernel<<<1, FEAT, 0, stream>>>(gcn, bn2_g, bn2_b, bn2h);

    // ---- heads: (16x512)@(512x128), mu/logvar straight into d_out ----
    gemm_nt_wmma<1><<<dim3(1, ZD / 16), 32, 0, stream>>>(bn2h, muwt, mu_b,
                                                         out + 16 * ZD, nullptr, ZD, FEAT, 0);
    gemm_nt_wmma<1><<<dim3(1, ZD / 16), 32, 0, stream>>>(bn2h, lvwt, lv_b,
                                                         out + 2 * 16 * ZD, nullptr, ZD, FEAT, 0);
    z_kernel<<<(16 * ZD + 255) / 256, 256, 0, stream>>>(out + 16 * ZD, out + 2 * 16 * ZD, eps, out);
}
